// Hierachical_70360154243242
// MI455X (gfx1250) — matrix operand order repaired and checked by reading, not compiled, not run on hardware
//
#include <hip/hip_runtime.h>

// ---------------------------------------------------------------------------
// CDNA5 (gfx1250) implementation of the hierarchical summarizer forward loss.
// All matmuls have M=16 (batch) -> V_WMMA_F32_16X16X32_F16, f32 accumulate.
// Weights are converted fp32->f16 once per launch; W_vocab (30.7MB f16) then
// lives in the 192MB L2 across the 114 decoder steps. GEMM waves own 4
// N-tiles so each A fragment feeds 4 WMMAs (1.25 fragment loads per WMMA).
// ---------------------------------------------------------------------------

typedef __attribute__((ext_vector_type(16))) _Float16 v16h;
typedef __attribute__((ext_vector_type(8)))  float    v8f;

#define DEVINL static __device__ __forceinline__

DEVINL float sigmf(float x) { return 1.0f / (1.0f + expf(-x)); }

DEVINL v8f wmma16(v16h a, v16h b, v8f c) {
  // D = A(16x32 f16) * B(32x16 f16) + C(16x16 f32)
  return __builtin_amdgcn_wmma_f32_16x16x32_f16(false, a, false, b,
                                                (short)0, c, false, false);
}

// A fragment: rows M=0..15 of row-major f16 A (leading dim lda), K block k0.
// ISA layout: lanes 0-15 hold K = k0+[0..7] (v0-3) and k0+[16..23] (v4-7);
// lanes 16-31 hold K = k0+[8..15] and k0+[24..31].
DEVINL v16h load_a16(const _Float16* A, int lda, int k0, int lane) {
  const _Float16* p = A + (size_t)(lane & 15) * lda + k0 + ((lane >> 4) << 3);
  v16h a;
#pragma unroll
  for (int j = 0; j < 8; ++j) { a[j] = p[j]; a[8 + j] = p[16 + j]; }
  return a;
}

// Same A fragment but gathered from an f32 row (per-lane row ptr), cvt to f16.
DEVINL v16h load_a32(const float* row, int k0, int lane) {
  const float* p = row + k0 + ((lane >> 4) << 3);
  v16h a;
#pragma unroll
  for (int j = 0; j < 8; ++j) {
    a[j]     = (_Float16)p[j];
    a[8 + j] = (_Float16)p[16 + j];
  }
  return a;
}

// B fragment: B[k][n] = W[n0+n][k0+k] from row-major f16 W [N,K] (ld = ldk).
// Lanes 0-15: n = lane, K = k0+[0..7],[16..23]; lanes 16-31: n = lane-16, K = k0+[8..15],[24..31].
DEVINL v16h load_b16(const _Float16* W, int ldk, int n0, int k0, int lane) {
  const _Float16* p = W + (size_t)(n0 + (lane & 15)) * ldk + k0 + ((lane >> 4) << 3);
  v16h b;
#pragma unroll
  for (int j = 0; j < 8; ++j) { b[j] = p[j]; b[8 + j] = p[16 + j]; }
  return b;
}

// ---------------------------------------------------------------------------
// fp32 -> f16 conversion
// ---------------------------------------------------------------------------
__global__ void cvt_f16_kernel(const float* __restrict__ src,
                               _Float16* __restrict__ dst, long n) {
  long i = (long)blockIdx.x * blockDim.x + threadIdx.x;
  if (i < n) dst[i] = (_Float16)src[i];
}

// ---------------------------------------------------------------------------
// Generic M=16 GEMM: C[16,N] = A[16,K](f16) @ W[N,K]^T(f16) + bias, f32 out.
// Each wave owns NT consecutive 16-col tiles, reusing one A fragment per
// K-block across NT WMMAs. Tile guards are wave-uniform (EXEC stays all-1s).
// Optional batching via blockIdx.y.
// ---------------------------------------------------------------------------
template <int NT>
__global__ __launch_bounds__(256) void gemm16_kernel(
    const _Float16* __restrict__ A, long strideA,
    const _Float16* __restrict__ W, const float* __restrict__ bias,
    float* __restrict__ C, long strideC, int N, int K) {
  int lane  = threadIdx.x & 31;
  int wave  = threadIdx.x >> 5;
  int tbase = (blockIdx.x * 8 + wave) * NT;  // first 16-col tile of this wave
  int ntiles = N >> 4;                       // N is a multiple of 16
  if (tbase >= ntiles) return;
  const _Float16* Ab = A + strideA * blockIdx.y;
  float* Cb = C + strideC * blockIdx.y;
  v8f acc[NT];
#pragma unroll
  for (int j = 0; j < NT; ++j) acc[j] = (v8f){};
  for (int k0 = 0; k0 < K; k0 += 32) {
    v16h a = load_a16(Ab, K, k0, lane);
#pragma unroll
    for (int j = 0; j < NT; ++j)
      if (tbase + j < ntiles)  // uniform per wave
        acc[j] = wmma16(a, load_b16(W, K, (tbase + j) << 4, k0, lane), acc[j]);
  }
  int rb = (lane >> 4) << 3;
#pragma unroll
  for (int j = 0; j < NT; ++j) {
    if (tbase + j >= ntiles) break;  // uniform per wave
    int col = ((tbase + j) << 4) + (lane & 15);
    float bv = bias ? bias[col] : 0.0f;
#pragma unroll
    for (int r = 0; r < 8; ++r) Cb[(size_t)(rb + r) * N + col] = acc[j][r] + bv;
  }
}

// ---------------------------------------------------------------------------
// Word encoder: one workgroup per article sentence. h (f16) and c (f32) live
// in LDS; each wave owns 4 hidden-unit tiles and computes all 4 gates fused.
// ---------------------------------------------------------------------------
__global__ __launch_bounds__(256) void word_enc_kernel(
    const int* __restrict__ articles, const float* __restrict__ emb,
    const _Float16* __restrict__ Wi, const _Float16* __restrict__ Wh,
    const float* __restrict__ bias, _Float16* __restrict__ out_hx) {
  const int H = 512, E = 512, L = 30;
  __shared__ float    c_s[16 * 512];
  __shared__ _Float16 h_s[16 * 512];
  __shared__ int tok[16];
  int s = blockIdx.x, tid = threadIdx.x;
  for (int i = tid; i < 16 * 512; i += 256) { c_s[i] = 0.0f; h_s[i] = (_Float16)0.0f; }
  __syncthreads();
  int lane = tid & 31, wave = tid >> 5;
  for (int t = 0; t < L; ++t) {
    if (tid < 16) tok[tid] = articles[(s * 16 + tid) * L + t];
    __syncthreads();
    const float* xrow = emb + (size_t)tok[lane & 15] * E;
    float hnew[4][8];
#pragma unroll
    for (int tl = 0; tl < 4; ++tl) {
      int j0 = (tl * 8 + wave) * 16;
      v8f a0 = {}, a1 = {}, a2 = {}, a3 = {};
      for (int k0 = 0; k0 < E; k0 += 32) {
        v16h a = load_a32(xrow, k0, lane);
        a0 = wmma16(a, load_b16(Wi, E, 0 * H + j0, k0, lane), a0);
        a1 = wmma16(a, load_b16(Wi, E, 1 * H + j0, k0, lane), a1);
        a2 = wmma16(a, load_b16(Wi, E, 2 * H + j0, k0, lane), a2);
        a3 = wmma16(a, load_b16(Wi, E, 3 * H + j0, k0, lane), a3);
      }
      for (int k0 = 0; k0 < H; k0 += 32) {
        v16h a = load_a16(h_s, H, k0, lane);
        a0 = wmma16(a, load_b16(Wh, H, 0 * H + j0, k0, lane), a0);
        a1 = wmma16(a, load_b16(Wh, H, 1 * H + j0, k0, lane), a1);
        a2 = wmma16(a, load_b16(Wh, H, 2 * H + j0, k0, lane), a2);
        a3 = wmma16(a, load_b16(Wh, H, 3 * H + j0, k0, lane), a3);
      }
      int col = j0 + (lane & 15);
      int rb  = (lane >> 4) << 3;
      float bi = bias[col], bf = bias[H + col], bg = bias[2 * H + col], bo = bias[3 * H + col];
#pragma unroll
      for (int r = 0; r < 8; ++r) {
        int row = rb + r;
        float iv = sigmf(a0[r] + bi), fv = sigmf(a1[r] + bf);
        float gv = tanhf(a2[r] + bg), ov = sigmf(a3[r] + bo);
        float cn = fv * c_s[row * H + col] + iv * gv;
        c_s[row * H + col] = cn;
        hnew[tl][r] = ov * tanhf(cn);
      }
    }
    __syncthreads();  // all waves done reading h before overwrite
#pragma unroll
    for (int tl = 0; tl < 4; ++tl) {
      int col = (tl * 8 + wave) * 16 + (lane & 15);
      int rb  = (lane >> 4) << 3;
#pragma unroll
      for (int r = 0; r < 8; ++r) h_s[(rb + r) * H + col] = (_Float16)hnew[tl][r];
    }
    __syncthreads();
  }
  for (int i = tid; i < 16 * 512; i += 256) out_hx[(size_t)s * 16 * 512 + i] = h_s[i];
}

// ---------------------------------------------------------------------------
// Sentence encoder: single workgroup, 20 steps over word_hx; emits sent_out
// per step and final (h,c) twice (decoder word state and sentence state).
// ---------------------------------------------------------------------------
__global__ __launch_bounds__(256) void sent_enc_kernel(
    const _Float16* __restrict__ in_hx, const _Float16* __restrict__ Wi,
    const _Float16* __restrict__ Wh, const float* __restrict__ bias,
    _Float16* __restrict__ sent_out, _Float16* __restrict__ w_h,
    float* __restrict__ w_c, _Float16* __restrict__ s_h, float* __restrict__ s_c) {
  const int H = 512, S = 20;
  __shared__ float    c_s[16 * 512];
  __shared__ _Float16 h_s[16 * 512];
  int tid = threadIdx.x;
  for (int i = tid; i < 16 * 512; i += 256) { c_s[i] = 0.0f; h_s[i] = (_Float16)0.0f; }
  __syncthreads();
  int lane = tid & 31, wave = tid >> 5;
  for (int t = 0; t < S; ++t) {
    const _Float16* x = in_hx + (size_t)t * 16 * H;
    float hnew[4][8];
#pragma unroll
    for (int tl = 0; tl < 4; ++tl) {
      int j0 = (tl * 8 + wave) * 16;
      v8f a0 = {}, a1 = {}, a2 = {}, a3 = {};
      for (int k0 = 0; k0 < H; k0 += 32) {
        v16h a = load_a16(x, H, k0, lane);
        a0 = wmma16(a, load_b16(Wi, H, 0 * H + j0, k0, lane), a0);
        a1 = wmma16(a, load_b16(Wi, H, 1 * H + j0, k0, lane), a1);
        a2 = wmma16(a, load_b16(Wi, H, 2 * H + j0, k0, lane), a2);
        a3 = wmma16(a, load_b16(Wi, H, 3 * H + j0, k0, lane), a3);
      }
      for (int k0 = 0; k0 < H; k0 += 32) {
        v16h a = load_a16(h_s, H, k0, lane);
        a0 = wmma16(a, load_b16(Wh, H, 0 * H + j0, k0, lane), a0);
        a1 = wmma16(a, load_b16(Wh, H, 1 * H + j0, k0, lane), a1);
        a2 = wmma16(a, load_b16(Wh, H, 2 * H + j0, k0, lane), a2);
        a3 = wmma16(a, load_b16(Wh, H, 3 * H + j0, k0, lane), a3);
      }
      int col = j0 + (lane & 15);
      int rb  = (lane >> 4) << 3;
      float bi = bias[col], bf = bias[H + col], bg = bias[2 * H + col], bo = bias[3 * H + col];
#pragma unroll
      for (int r = 0; r < 8; ++r) {
        int row = rb + r;
        float iv = sigmf(a0[r] + bi), fv = sigmf(a1[r] + bf);
        float gv = tanhf(a2[r] + bg), ov = sigmf(a3[r] + bo);
        float cn = fv * c_s[row * H + col] + iv * gv;
        c_s[row * H + col] = cn;
        hnew[tl][r] = ov * tanhf(cn);
      }
    }
    __syncthreads();
#pragma unroll
    for (int tl = 0; tl < 4; ++tl) {
      int col = (tl * 8 + wave) * 16 + (lane & 15);
      int rb  = (lane >> 4) << 3;
#pragma unroll
      for (int r = 0; r < 8; ++r) h_s[(rb + r) * H + col] = (_Float16)hnew[tl][r];
    }
    __syncthreads();
    for (int i = tid; i < 16 * 512; i += 256)
      sent_out[(size_t)t * 16 * 512 + i] = h_s[i];
    __syncthreads();
  }
  for (int i = tid; i < 16 * 512; i += 256) {
    w_h[i] = h_s[i]; s_h[i] = h_s[i];
    w_c[i] = c_s[i]; s_c[i] = c_s[i];
  }
}

// ---------------------------------------------------------------------------
// Single LSTM step, global h(f16)/c(f32) state updated in place (1 workgroup).
// ---------------------------------------------------------------------------
__global__ __launch_bounds__(256) void lstm_step_kernel(
    const _Float16* __restrict__ x, int K, const _Float16* __restrict__ Wi,
    const _Float16* __restrict__ Wh, const float* __restrict__ bias,
    _Float16* __restrict__ h, float* __restrict__ c) {
  const int H = 512;
  int lane = threadIdx.x & 31, wave = threadIdx.x >> 5;
  float hnew[4][8];
#pragma unroll
  for (int tl = 0; tl < 4; ++tl) {
    int j0 = (tl * 8 + wave) * 16;
    v8f a0 = {}, a1 = {}, a2 = {}, a3 = {};
    for (int k0 = 0; k0 < K; k0 += 32) {
      v16h a = load_a16(x, K, k0, lane);
      a0 = wmma16(a, load_b16(Wi, K, 0 * H + j0, k0, lane), a0);
      a1 = wmma16(a, load_b16(Wi, K, 1 * H + j0, k0, lane), a1);
      a2 = wmma16(a, load_b16(Wi, K, 2 * H + j0, k0, lane), a2);
      a3 = wmma16(a, load_b16(Wi, K, 3 * H + j0, k0, lane), a3);
    }
    for (int k0 = 0; k0 < H; k0 += 32) {
      v16h a = load_a16(h, H, k0, lane);
      a0 = wmma16(a, load_b16(Wh, H, 0 * H + j0, k0, lane), a0);
      a1 = wmma16(a, load_b16(Wh, H, 1 * H + j0, k0, lane), a1);
      a2 = wmma16(a, load_b16(Wh, H, 2 * H + j0, k0, lane), a2);
      a3 = wmma16(a, load_b16(Wh, H, 3 * H + j0, k0, lane), a3);
    }
    int col = j0 + (lane & 15);
    int rb  = (lane >> 4) << 3;
    float bi = bias[col], bf = bias[H + col], bg = bias[2 * H + col], bo = bias[3 * H + col];
#pragma unroll
    for (int r = 0; r < 8; ++r) {
      int row = rb + r;
      float iv = sigmf(a0[r] + bi), fv = sigmf(a1[r] + bf);
      float gv = tanhf(a2[r] + bg), ov = sigmf(a3[r] + bo);
      float cn = fv * c[row * H + col] + iv * gv;
      c[row * H + col] = cn;
      hnew[tl][r] = ov * tanhf(cn);
    }
  }
  __syncthreads();  // all waves finished reading h
#pragma unroll
  for (int tl = 0; tl < 4; ++tl) {
    int col = (tl * 8 + wave) * 16 + (lane & 15);
    int rb  = (lane >> 4) << 3;
#pragma unroll
    for (int r = 0; r < 8; ++r) h[(rb + r) * H + col] = (_Float16)hnew[tl][r];
  }
}

// ---------------------------------------------------------------------------
// Decoder input embedding gather (f32 -> f16)
// ---------------------------------------------------------------------------
__global__ void embed_tgt_kernel(const float* __restrict__ emb,
                                 const int* __restrict__ summaries, int n, int t,
                                 _Float16* __restrict__ x) {
  const int T = 20, E = 512;
  int i = blockIdx.x * 256 + threadIdx.x;
  if (i >= 16 * E) return;
  int b = i / E, hh = i % E;
  int wb = summaries[(n * 16 + b) * T + t];
  x[i] = (_Float16)emb[(size_t)wb * E + hh];
}

// ---------------------------------------------------------------------------
// Per-row masked NLL over 30000-way logits (two-pass log-softmax)
// ---------------------------------------------------------------------------
__global__ __launch_bounds__(256) void nll_rows_kernel(
    const float* __restrict__ logits, const int* __restrict__ summaries,
    int n, int t, float* __restrict__ nll, float* __restrict__ msk) {
  const int V = 30000, T = 20;
  int b = blockIdx.x;
  const float* row = logits + (size_t)b * V;
  __shared__ float red[256];
  float mx = -3.4e38f;
  for (int i = threadIdx.x; i < V; i += 256) mx = fmaxf(mx, row[i]);
  red[threadIdx.x] = mx; __syncthreads();
  for (int s = 128; s > 0; s >>= 1) {
    if (threadIdx.x < s) red[threadIdx.x] = fmaxf(red[threadIdx.x], red[threadIdx.x + s]);
    __syncthreads();
  }
  mx = red[0]; __syncthreads();
  float se = 0.0f;
  for (int i = threadIdx.x; i < V; i += 256) se += expf(row[i] - mx);
  red[threadIdx.x] = se; __syncthreads();
  for (int s = 128; s > 0; s >>= 1) {
    if (threadIdx.x < s) red[threadIdx.x] += red[threadIdx.x + s];
    __syncthreads();
  }
  if (threadIdx.x == 0) {
    int wa = summaries[(n * 16 + b) * T + t + 1];
    nll[b] = logf(red[0]) + mx - row[wa];
    msk[b] = (wa != 0) ? 1.0f : 0.0f;
  }
}

__global__ void loss_word_kernel(const float* __restrict__ nll,
                                 const float* __restrict__ msk,
                                 float* __restrict__ loss) {
  if (threadIdx.x == 0) {
    float s = 0.0f, sm = 0.0f;
    for (int b = 0; b < 16; ++b) { s += nll[b] * msk[b]; sm += msk[b]; }
    loss[0] += s / fmaxf(sm, 1.0f);
  }
}

// ---------------------------------------------------------------------------
// Attention energies: e[s][b] = sum_h tanh(sent_feat + q + cov*wc) * v, masked
// ---------------------------------------------------------------------------
__global__ __launch_bounds__(256) void attn_e_kernel(
    const float* __restrict__ sent_feat, const float* __restrict__ q,
    const float* __restrict__ cov, const float* __restrict__ wc,
    const float* __restrict__ vvec, const float* __restrict__ masksb,
    float* __restrict__ e) {
  const int H = 512;
  int s = blockIdx.x;
  __shared__ float red[256];
  for (int b = 0; b < 16; ++b) {
    float p = 0.0f;
    float cv = cov[s * 16 + b];
    for (int hh = threadIdx.x; hh < H; hh += 256)
      p += tanhf(sent_feat[((size_t)s * 16 + b) * H + hh] + q[b * H + hh] + cv * wc[hh]) * vvec[hh];
    red[threadIdx.x] = p; __syncthreads();
    for (int st = 128; st > 0; st >>= 1) {
      if (threadIdx.x < st) red[threadIdx.x] += red[threadIdx.x + st];
      __syncthreads();
    }
    if (threadIdx.x == 0)
      e[s * 16 + b] = (masksb[s * 16 + b] > 0.0f) ? red[0] : -1e9f;
    __syncthreads();
  }
}

// softmax over S per batch row + coverage loss (pre-update) + coverage update
__global__ void softmax_cov_kernel(const float* __restrict__ e,
                                   float* __restrict__ align,
                                   float* __restrict__ cov,
                                   float* __restrict__ loss) {
  const int S = 20;
  __shared__ float cl[16];
  int b = threadIdx.x;
  if (b < 16) {
    float mx = -3.4e38f;
    for (int s = 0; s < S; ++s) mx = fmaxf(mx, e[s * 16 + b]);
    float den = 0.0f;
    for (int s = 0; s < S; ++s) den += expf(e[s * 16 + b] - mx);
    float covl = 0.0f;
    for (int s = 0; s < S; ++s) {
      float al = expf(e[s * 16 + b] - mx) / den;
      align[s * 16 + b] = al;
      covl += fminf(al, cov[s * 16 + b]);
      cov[s * 16 + b] += al;
    }
    cl[b] = covl;
  }
  __syncthreads();
  if (threadIdx.x == 0) {
    float s = 0.0f;
    for (int b2 = 0; b2 < 16; ++b2) s += cl[b2];
    loss[0] += s / 16.0f;
  }
}

// context = sum_s align*sent_out ; build concat = [context, s_h2] in f16
__global__ void context_kernel(const float* __restrict__ align,
                               const _Float16* __restrict__ sent_out,
                               const _Float16* __restrict__ s_h2,
                               _Float16* __restrict__ concat) {
  const int S = 20, H = 512;
  int i = blockIdx.x * 256 + threadIdx.x;
  if (i >= 16 * H) return;
  int b = i / H, hh = i % H;
  float ctx = 0.0f;
  for (int s = 0; s < S; ++s)
    ctx += align[s * 16 + b] * (float)sent_out[((size_t)s * 16 + b) * H + hh];
  concat[b * 1024 + hh]       = (_Float16)ctx;
  concat[b * 1024 + 512 + hh] = s_h2[b * H + hh];
}

// final = tanh(gout); becomes both next w_h and next s_h (f16)
__global__ void tanh_final_kernel(const float* __restrict__ gout,
                                  _Float16* __restrict__ w_h,
                                  _Float16* __restrict__ s_h) {
  int i = blockIdx.x * 256 + threadIdx.x;
  if (i >= 16 * 512) return;
  _Float16 v = (_Float16)tanhf(gout[i]);
  w_h[i] = v; s_h[i] = v;
}

// zero loss + coverage, build attention mask from articles[:,:,0]
__global__ void init_kernel(float* __restrict__ cov, float* __restrict__ masksb,
                            const int* __restrict__ articles,
                            float* __restrict__ loss) {
  int i = blockIdx.x * 256 + threadIdx.x;
  if (i == 0) loss[0] = 0.0f;
  if (i < 20 * 16) {
    cov[i] = 0.0f;
    int s = i / 16, b = i % 16;
    masksb[i] = (articles[(s * 16 + b) * 30] > 0) ? 1.0f : 0.0f;
  }
}

// ---------------------------------------------------------------------------
extern "C" void kernel_launch(void* const* d_in, const int* in_sizes, int n_in,
                              void* d_out, int out_size, void* d_ws, size_t ws_size,
                              hipStream_t stream) {
  (void)in_sizes; (void)n_in; (void)out_size; (void)ws_size;
  const int S = 20, Nn = 6, T = 20, H = 512, E = 512, Vt = 30000;

  const int*   articles  = (const int*)d_in[0];
  const int*   summaries = (const int*)d_in[1];
  const float* emb_src   = (const float*)d_in[2];
  const float* emb_tgt   = (const float*)d_in[3];
  const float* we_Wi = (const float*)d_in[4];
  const float* we_Wh = (const float*)d_in[5];
  const float* we_b  = (const float*)d_in[6];
  const float* se_Wi = (const float*)d_in[7];
  const float* se_Wh = (const float*)d_in[8];
  const float* se_b  = (const float*)d_in[9];
  const float* W_feat = (const float*)d_in[10];
  const float* b_feat = (const float*)d_in[11];
  const float* wd_Wi = (const float*)d_in[12];
  const float* wd_Wh = (const float*)d_in[13];
  const float* wd_b  = (const float*)d_in[14];
  const float* W_vocab = (const float*)d_in[15];
  const float* b_vocab = (const float*)d_in[16];
  const float* sd_Wi = (const float*)d_in[17];
  const float* sd_Wh = (const float*)d_in[18];
  const float* sd_b  = (const float*)d_in[19];
  const float* attn_Wh = (const float*)d_in[20];
  const float* attn_wc = (const float*)d_in[21];
  const float* attn_v  = (const float*)d_in[22];
  const float* W_out = (const float*)d_in[23];
  const float* b_out = (const float*)d_in[24];

  float* loss = (float*)d_out;  // scalar loss accumulated in place

  // ---- bump allocator over workspace (~55 MB) ----
  char*  ws  = (char*)d_ws;
  size_t off = 0;
  auto a16 = [&](size_t elems) -> _Float16* {
    void* p = ws + off; off = (off + elems * 2 + 255) & ~(size_t)255; return (_Float16*)p;
  };
  auto a32 = [&](size_t elems) -> float* {
    void* p = ws + off; off = (off + elems * 4 + 255) & ~(size_t)255; return (float*)p;
  };

  _Float16* weWi16 = a16((size_t)4 * H * E);
  _Float16* weWh16 = a16((size_t)4 * H * H);
  _Float16* seWi16 = a16((size_t)4 * H * H);
  _Float16* seWh16 = a16((size_t)4 * H * H);
  _Float16* wdWi16 = a16((size_t)4 * H * E);
  _Float16* wdWh16 = a16((size_t)4 * H * H);
  _Float16* sdWi16 = a16((size_t)4 * H * H);
  _Float16* sdWh16 = a16((size_t)4 * H * H);
  _Float16* Wv16    = a16((size_t)Vt * H);
  _Float16* Wfeat16 = a16((size_t)H * H);
  _Float16* aWh16   = a16((size_t)H * H);
  _Float16* Wout16  = a16((size_t)H * 2 * H);
  _Float16* word_hx  = a16((size_t)S * 16 * H);
  _Float16* sent_out = a16((size_t)S * 16 * H);
  _Float16* x16      = a16((size_t)16 * E);
  _Float16* w_h      = a16((size_t)16 * H);
  _Float16* s_h      = a16((size_t)16 * H);
  _Float16* concat16 = a16((size_t)16 * 2 * H);
  float* sent_feat = a32((size_t)S * 16 * H);
  float* w_c    = a32((size_t)16 * H);
  float* s_c    = a32((size_t)16 * H);
  float* logits = a32((size_t)16 * Vt);
  float* q      = a32((size_t)16 * H);
  float* gout   = a32((size_t)16 * H);
  float* e_buf  = a32((size_t)S * 16);
  float* align  = a32((size_t)S * 16);
  float* cov    = a32((size_t)S * 16);
  float* masksb = a32((size_t)S * 16);
  float* nllb   = a32(16);
  float* mskb   = a32(16);

  auto cvt = [&](const float* src, _Float16* dst, long n) {
    cvt_f16_kernel<<<(int)((n + 255) / 256), 256, 0, stream>>>(src, dst, n);
  };
  cvt(we_Wi, weWi16, (long)4 * H * E);  cvt(we_Wh, weWh16, (long)4 * H * H);
  cvt(se_Wi, seWi16, (long)4 * H * H);  cvt(se_Wh, seWh16, (long)4 * H * H);
  cvt(wd_Wi, wdWi16, (long)4 * H * E);  cvt(wd_Wh, wdWh16, (long)4 * H * H);
  cvt(sd_Wi, sdWi16, (long)4 * H * H);  cvt(sd_Wh, sdWh16, (long)4 * H * H);
  cvt(W_vocab, Wv16, (long)Vt * H);
  cvt(W_feat, Wfeat16, (long)H * H);
  cvt(attn_Wh, aWh16, (long)H * H);
  cvt(W_out, Wout16, (long)H * 2 * H);

  init_kernel<<<2, 256, 0, stream>>>(cov, masksb, articles, loss);

  // ---- encoders ----
  word_enc_kernel<<<S, 256, 0, stream>>>(articles, emb_src, weWi16, weWh16, we_b, word_hx);
  sent_enc_kernel<<<1, 256, 0, stream>>>(word_hx, seWi16, seWh16, se_b,
                                         sent_out, w_h, w_c, s_h, s_c);
  // sent_feat[s] = sent_out[s] @ W_feat^T + b_feat   (batched over S)
  // N=512 -> 32 tiles = 1 block at 8 waves x 4 tiles
  gemm16_kernel<4><<<dim3(1, S), 256, 0, stream>>>(sent_out, (long)16 * H, Wfeat16,
                                                   b_feat, sent_feat, (long)16 * H, H, H);

  // ---- hierarchical decoder ----
  // vocab: 1875 tiles -> 469 wave-groups of 4 tiles -> 59 blocks of 8 waves
  const int vocab_groups = (Vt / 16 + 3) / 4;
  const int vocab_blocks = (vocab_groups + 7) / 8;
  for (int n = 0; n < Nn; ++n) {
    for (int t = 0; t < T - 1; ++t) {
      embed_tgt_kernel<<<32, 256, 0, stream>>>(emb_tgt, summaries, n, t, x16);
      lstm_step_kernel<<<1, 256, 0, stream>>>(x16, E, wdWi16, wdWh16, wd_b, w_h, w_c);
      gemm16_kernel<4><<<dim3(vocab_blocks, 1), 256, 0, stream>>>(
          w_h, 0, Wv16, b_vocab, logits, 0, Vt, H);
      nll_rows_kernel<<<16, 256, 0, stream>>>(logits, summaries, n, t, nllb, mskb);
      loss_word_kernel<<<1, 32, 0, stream>>>(nllb, mskb, loss);
    }
    // sentence decoder step: s_h <- s_h2
    lstm_step_kernel<<<1, 256, 0, stream>>>(w_h, H, sdWi16, sdWh16, sd_b, s_h, s_c);
    gemm16_kernel<4><<<dim3(1, 1), 256, 0, stream>>>(s_h, 0, aWh16, nullptr, q, 0, H, H);
    attn_e_kernel<<<S, 256, 0, stream>>>(sent_feat, q, cov, attn_wc, attn_v, masksb, e_buf);
    softmax_cov_kernel<<<1, 32, 0, stream>>>(e_buf, align, cov, loss);
    context_kernel<<<32, 256, 0, stream>>>(align, sent_out, s_h, concat16);
    gemm16_kernel<4><<<dim3(1, 1), 256, 0, stream>>>(concat16, 0, Wout16, b_out, gout, 0, H, 2 * H);
    tanh_final_kernel<<<32, 256, 0, stream>>>(gout, w_h, s_h);
  }
}
